// SparseConv3d_29970281792092
// MI455X (gfx1250) — matrix operand order repaired and checked by reading, not compiled, not run on hardware
//
#include <hip/hip_runtime.h>
#include <cstdint>

// CDNA5 / gfx1250, wave32. Implicit-GEMM 3x3x3 conv via v_wmma_f32_16x16x32_bf16.
typedef __attribute__((ext_vector_type(16))) __bf16 v16bf;
typedef __attribute__((ext_vector_type(8)))  float  v8f;

__device__ __forceinline__ unsigned f2bf(float f) {
  // fp32 -> bf16 round-to-nearest-even (low 16 bits of result)
  unsigned u = __float_as_uint(f);
  unsigned r = ((u >> 16) & 1u) + 0x7FFFu;
  return ((u + r) >> 16) & 0xFFFFu;
}

// ---------------------------------------------------------------------------
// Kernel 0: repack weights (Cout=64, Cin=32, 3,3,3) fp32 -> bf16 pair words in
// d_ws with layout word[(tap*64 + co)*16 + cp] = {ci=2cp (lo), ci=2cp+1 (hi)}.
// 27*64*16 = 27648 words = 108 KiB.
// ---------------------------------------------------------------------------
__global__ void repack_weights_k(const float* __restrict__ w,
                                 unsigned int* __restrict__ wsw) {
  int t = blockIdx.x * 256 + threadIdx.x;     // 0 .. 27647
  int cp  = t & 15;
  int co  = (t >> 4) & 63;
  int tap = t >> 10;                          // 0 .. 26  (kz*9 + ky*3 + kx)
  size_t wi = ((size_t)(co * 32 + cp * 2)) * 27 + tap;
  unsigned lo = f2bf(w[wi]);
  unsigned hi = f2bf(w[wi + 27]);             // next ci
  wsw[t] = lo | (hi << 16);
}

// ---------------------------------------------------------------------------
// Main kernel: one 256-thread block (8 wave32) per output row (b, z, y).
// LDS holds the 3x3 (dz,dy) halo of the row for all 32 input channels,
// bf16 ci-pair packed: word[(row*66 + xl)*20 + cp], xl in 0..65 (gx = xl-1).
// Wave w: co_tile = w&3 (16 Cout), x_tiles = {2*(w>>2), 2*(w>>2)+1}.
// 27 taps -> 27 WMMA (K = Cin = 32) per 16x16 tile.
// ---------------------------------------------------------------------------
__global__ __launch_bounds__(256)
void sparse_conv3d_wmma_k(const float* __restrict__ xin,
                          const unsigned char* __restrict__ mask, // jnp.bool_: 1 byte
                          const float* __restrict__ bias,
                          const unsigned int* __restrict__ wsw,
                          float* __restrict__ out) {
  constexpr int XLP = 66;   // xl extent
  constexpr int CPW = 20;   // cp stride in words (16 used + 4 pad vs bank conflicts)
  __shared__ unsigned int lds_in[9 * XLP * CPW];   // 47520 B

  const int tid = threadIdx.x;
  const int blk = blockIdx.x;                 // 0 .. 8191
  const int b   = blk >> 12;                  // /4096
  const int z0  = (blk >> 6) & 63;
  const int y0  = blk & 63;

  // ---- stage input halo (fp32 -> packed bf16 pairs) ----------------------
  auto stage = [&](int row, int cp, int xl) {
    int gz = z0 + row / 3 - 1;
    int gy = y0 + row % 3 - 1;
    int gx = xl - 1;
    unsigned v = 0;
    if ((unsigned)gz < 64u && (unsigned)gy < 64u && (unsigned)gx < 64u) {
      size_t base = ((size_t)(b * 32 + cp * 2)) * 262144 +
                    (size_t)gz * 4096 + gy * 64 + gx;
      v = f2bf(xin[base]) | (f2bf(xin[base + 262144]) << 16);  // ci, ci+1
    }
    lds_in[(row * XLP + xl) * CPW + cp] = v;
  };

  // main body: xl 0..63, 144 (row,cp) combos, 4 combos per pass
  for (int p = 0; p < 36; ++p) {
    int combo = p * 4 + (tid >> 6);
    stage(combo >> 4, combo & 15, tid & 63);
  }
  // tail: xl 64,65 for all 144 combos (288 elems)
  for (int p = 0; p < 2; ++p) {
    int idx = p * 256 + tid;
    if (idx < 288) {
      int combo = idx >> 1;
      stage(combo >> 4, combo & 15, 64 + (idx & 1));
    }
  }
  __syncthreads();

  // ---- per-wave WMMA accumulation ---------------------------------------
  const int wid  = tid >> 5;
  const int lane = tid & 31;
  const int hh   = lane >> 4;     // half-wave select
  const int ln   = lane & 15;
  const int co0  = (wid & 3) * 16;
  const int xt0  = (wid >> 2) * 2;

  v8f acc0 = {};
  v8f acc1 = {};
  const uint4* wsw4 = (const uint4*)wsw;

  for (int row = 0; row < 9; ++row) {
#pragma unroll
    for (int kx = 0; kx < 3; ++kx) {
      const int tap = row * 3 + kx;
      // A fragment (16x32 bf16, §7.12.2): lane M = ln; dwords 0..3 at
      // cp = hh*4 + v, dwords 4..7 at cp = hh*4 + 8 + v  -> two b128 loads.
      const uint4* pa = wsw4 + (size_t)(tap * 64 + co0 + ln) * 4 + hh;
      union { uint4 q[2]; v16bf v; } A;
      A.q[0] = pa[0];
      A.q[1] = pa[2];

      // B fragment (32x16 bf16): lane N = ln; dwords 0..3 at
      // cp = hh*4 + v, dwords 4..7 at cp = hh*4 + 8 + v  -> two ds_load_b128.
      {
        int xl = xt0 * 16 + ln + kx;
        const uint4* pb = (const uint4*)&lds_in[(row * XLP + xl) * CPW + hh * 4];
        union { uint4 q[2]; v16bf v; } Bf;
        Bf.q[0] = pb[0];
        Bf.q[1] = pb[2];
        acc0 = __builtin_amdgcn_wmma_f32_16x16x32_bf16(
            false, A.v, false, Bf.v, (short)0, acc0, false, false);
      }
      {
        int xl = (xt0 + 1) * 16 + ln + kx;
        const uint4* pb = (const uint4*)&lds_in[(row * XLP + xl) * CPW + hh * 4];
        union { uint4 q[2]; v16bf v; } Bf;
        Bf.q[0] = pb[0];
        Bf.q[1] = pb[2];
        acc1 = __builtin_amdgcn_wmma_f32_16x16x32_bf16(
            false, A.v, false, Bf.v, (short)0, acc1, false, false);
      }
    }
  }

  // ---- epilogue: bias add, mask, store (C/D layout §7.12.2) -------------
  const float4* bp = (const float4*)(bias + co0 + hh * 8);
  float4 bv0 = bp[0];
  float4 bv1 = bp[1];
  float bb[8] = {bv0.x, bv0.y, bv0.z, bv0.w, bv1.x, bv1.y, bv1.z, bv1.w};

#pragma unroll
  for (int t = 0; t < 2; ++t) {
    const v8f acc = (t == 0) ? acc0 : acc1;
    const int x  = (xt0 + t) * 16 + ln;           // C/D: N = lane&15
    const int sp = z0 * 4096 + y0 * 64 + x;
    const float mf = mask[(size_t)b * 262144 + sp] ? 1.0f : 0.0f;
#pragma unroll
    for (int r = 0; r < 8; ++r) {                 // C/D: M = r + hh*8
      const int co = co0 + hh * 8 + r;
      out[((size_t)(b * 64 + co)) * 262144 + sp] = (acc[r] + bb[r]) * mf;
    }
  }
}

// ---------------------------------------------------------------------------
extern "C" void kernel_launch(void* const* d_in, const int* in_sizes, int n_in,
                              void* d_out, int out_size, void* d_ws, size_t ws_size,
                              hipStream_t stream) {
  const float*         x    = (const float*)d_in[0];
  const unsigned char* mask = (const unsigned char*)d_in[1];  // jnp.bool_ = 1 byte
  const float*         w    = (const float*)d_in[2];
  const float*         bias = (const float*)d_in[3];
  float*               out  = (float*)d_out;
  unsigned int*        wsw  = (unsigned int*)d_ws;            // needs 108 KiB

  // Repack weights into WMMA A-fragment layout (same stream -> ordered).
  repack_weights_k<<<108, 256, 0, stream>>>(w, wsw);

  // One block per (b, z, y) output row: 2*64*64 = 8192 blocks.
  sparse_conv3d_wmma_k<<<8192, 256, 0, stream>>>(x, mask, bias, wsw, out);
}
